// WeightOnlyLinear_85014582657685
// MI455X (gfx1250) — compile-verified, matrix operand order checked by reading, not run on hardware
//
#include <hip/hip_runtime.h>
#include <hip/hip_bf16.h>

typedef __attribute__((ext_vector_type(16))) __bf16 v16bf;
typedef __attribute__((ext_vector_type(8)))  __bf16 v8bf;
typedef __attribute__((ext_vector_type(4)))  __bf16 v4bf;
typedef __attribute__((ext_vector_type(8)))  float  v8f;

union FragU { v16bf v; v8bf h[2]; };

// Tile config
#define BM 128
#define BN 128
#define BK 64
#define LDT 72   // LDS row stride in bf16 elements (144B: 16B-aligned, conflict-padded)

__global__ __launch_bounds__(256, 1)
void gptq_wmma_bf16_kernel(const float* __restrict__ x,
                           const int*   __restrict__ qweight,
                           const int*   __restrict__ qzeros,
                           const float* __restrict__ scales,
                           const float* __restrict__ bias,
                           float* __restrict__ out,
                           int M, int N, int K)
{
    __shared__ __bf16 As[BM * LDT];   // x tile, [row m][k], bf16
    __shared__ __bf16 Bs[BN * LDT];   // dequant W^T tile, [col n][k], bf16

    const int t    = threadIdx.x;
    const int lane = t & 31;
    const int wave = t >> 5;

    const int bm0 = blockIdx.y * BM;
    const int bn0 = blockIdx.x * BN;

    // wave -> 64x32 sub-tile of the 128x128 block (2 waves in M, 4 in N)
    const int wm = (wave >> 2) * 64;
    const int wn = (wave & 3)  * 32;

    // A-tile loader assignment: 16 float4 per row, 16 rows/pass, 8 passes
    const int acol  = (t & 15) * 4;   // k offset within BK (floats)
    const int arow0 = t >> 4;         // 0..15

    // B-tile loader assignment: 2 threads per column (4 packed words each)
    const int nloc   = t & 127;
    const int k8base = (t >> 7) * 4;  // 0 or 4 (of 8 words covering BK=64)
    const int gn_b   = bn0 + nloc;

    // fragment-load lane geometry (16-bit A/B 16x32 layout)
    const int r  = lane & 15;
    const int kh = (lane >> 4) * 8;   // K half select within 16-elem group

    v8f acc[4][2];
    #pragma unroll
    for (int mt = 0; mt < 4; ++mt)
        #pragma unroll
        for (int nt = 0; nt < 2; ++nt)
            acc[mt][nt] = (v8f){0.f,0.f,0.f,0.f,0.f,0.f,0.f,0.f};

    const int nwords = N >> 3;   // qzeros row length

    for (int k0 = 0; k0 < K; k0 += BK) {
        // ---- stage A tile: fp32 -> bf16 ----
        #pragma unroll
        for (int p = 0; p < 8; ++p) {
            const int row = arow0 + p * 16;
            const float4 f = *(const float4*)(x + (size_t)(bm0 + row) * K + k0 + acol);
            v4bf h;
            h[0] = (__bf16)f.x; h[1] = (__bf16)f.y;
            h[2] = (__bf16)f.z; h[3] = (__bf16)f.w;
            *(v4bf*)&As[row * LDT + acol] = h;
        }

        // ---- stage B tile: int4 dequant -> bf16 (GROUPSIZE=128 >= BK, one group/step) ----
        const int g = k0 >> 7;
        const int zword = qzeros[(size_t)g * nwords + (gn_b >> 3)];
        const int zp = ((zword >> ((gn_b & 7) * 4)) & 15) + 1;
        const float sc = scales[(size_t)g * N + gn_b];
        const float zs = -(float)zp * sc;

        #pragma unroll
        for (int i = 0; i < 4; ++i) {
            const int k8 = k8base + i;
            const unsigned w =
                (unsigned)qweight[(size_t)((k0 >> 3) + k8) * N + gn_b];
            v8bf q;
            #pragma unroll
            for (int j = 0; j < 8; ++j) {
                const float nib = (float)((w >> (4 * j)) & 15u);
                q[j] = (__bf16)fmaf(nib, sc, zs);
            }
            *(v8bf*)&Bs[nloc * LDT + k8 * 8] = q;
        }

        __syncthreads();

        // ---- WMMA over the staged tiles ----
        #pragma unroll
        for (int ko = 0; ko < BK; ko += 32) {
            v16bf af[4], bf[2];
            #pragma unroll
            for (int mt = 0; mt < 4; ++mt) {
                const __bf16* p = &As[(wm + mt * 16 + r) * LDT + ko + kh];
                FragU u;
                u.h[0] = *(const v8bf*)p;
                u.h[1] = *(const v8bf*)(p + 16);
                af[mt] = u.v;
            }
            #pragma unroll
            for (int nt = 0; nt < 2; ++nt) {
                const __bf16* p = &Bs[(wn + nt * 16 + r) * LDT + ko + kh];
                FragU u;
                u.h[0] = *(const v8bf*)p;
                u.h[1] = *(const v8bf*)(p + 16);
                bf[nt] = u.v;
            }
            #pragma unroll
            for (int mt = 0; mt < 4; ++mt)
                #pragma unroll
                for (int nt = 0; nt < 2; ++nt)
                    acc[mt][nt] = __builtin_amdgcn_wmma_f32_16x16x32_bf16(
                        false, af[mt], false, bf[nt],
                        (short)0, acc[mt][nt], false, false);
        }

        __syncthreads();
    }

    // ---- epilogue: C/D layout -> global fp32 + bias ----
    const int rowHalf = (lane >> 4) * 8;
    #pragma unroll
    for (int nt = 0; nt < 2; ++nt) {
        const int gn = bn0 + wn + nt * 16 + r;
        const float bv = bias[gn];
        #pragma unroll
        for (int mt = 0; mt < 4; ++mt) {
            const int gmBase = bm0 + wm + mt * 16 + rowHalf;
            #pragma unroll
            for (int v = 0; v < 8; ++v) {
                out[(size_t)(gmBase + v) * N + gn] = acc[mt][nt][v] + bv;
            }
        }
    }
}

extern "C" void kernel_launch(void* const* d_in, const int* in_sizes, int n_in,
                              void* d_out, int out_size, void* d_ws, size_t ws_size,
                              hipStream_t stream) {
    const float* x       = (const float*)d_in[0];
    const int*   qweight = (const int*)  d_in[1];
    const int*   qzeros  = (const int*)  d_in[2];
    const float* scales  = (const float*)d_in[3];
    const float* bias    = (const float*)d_in[4];
    float* out = (float*)d_out;

    const int N = in_sizes[4];                              // out_f (11008)
    const int K = (int)(((long long)in_sizes[1] * 8) / N);  // in_f  (4096)
    const int M = in_sizes[0] / K;                          // tokens (4096)

    dim3 grid(N / BN, M / BM);
    gptq_wmma_bf16_kernel<<<grid, 256, 0, stream>>>(
        x, qweight, qzeros, scales, bias, out, M, N, K);
}
